// ContinuousConv1d_40192303956162
// MI455X (gfx1250) — matrix operand order repaired and checked by reading, not compiled, not run on hardware
//
#include <hip/hip_runtime.h>
#include <hip/hip_bf16.h>

typedef __attribute__((ext_vector_type(16))) _Float16 v16h;
typedef __attribute__((ext_vector_type(8)))  float    v8f;

#define KSZ   64
#define NFFT  64
#define IC    32
#define OC    64
#define BN    4
#define T1N   4096
#define T2N   512
#define TTILE 16
#define XTL   4176   // 32 zero-pad + 4096 + tail pad (mult of 8 halves, covers sa+72)

union HU { unsigned u; _Float16 h[2]; };

// ---------------------------------------------------------------------------
// Prep A: pack kernel[o][c][m] (f32) into f16 WMMA-B fragment order so the
// stage-4 B operand is two consecutive global_load_b128 per lane.
//   Wh[((otile*64 + kcg)*32 + lane)*8 + j] packs halves for
//   N = otile*16 + lane%16, K = (lane>=16?16:0) + 2j + {0,1}; m = kcg, c = K.
// ---------------------------------------------------------------------------
__global__ __launch_bounds__(256) void cconv_prep_w(const float* __restrict__ kern,
                                                    unsigned* __restrict__ Wh) {
  int gid   = blockIdx.x * 256 + threadIdx.x;   // < 4*64*32*8 = 65536
  int j     = gid & 7;
  int lane  = (gid >> 3) & 31;
  int kcg   = (gid >> 8) & 63;
  int otile = gid >> 14;
  int o = otile * 16 + (lane & 15);
  int c = ((lane >> 4) << 4) + 2 * j;
  int m = kcg;
  HU cv;
  cv.h[0] = (_Float16)kern[((size_t)(o * IC + c)) * NFFT + m];
  cv.h[1] = (_Float16)kern[((size_t)(o * IC + c + 1)) * NFFT + m];
  Wh[gid] = cv.u;
}

// ---------------------------------------------------------------------------
// Prep B: zero-padded channel-major f16 image of x:
//   xt[b][c][l] = x[b][l-32][c]  (0 outside [32, 32+T1)), l < XTL.
// Windows become contiguous runs along l -> perfect for async b128 LDS copies.
// ---------------------------------------------------------------------------
__global__ __launch_bounds__(256) void cconv_prep_x(const float* __restrict__ x,
                                                    _Float16* __restrict__ xt) {
  int gid = blockIdx.x * 256 + threadIdx.x;     // < 4*32*4176 = 534528
  int l   = gid % XTL;
  int bc  = gid / XTL;
  int c   = bc & 31;
  int b   = bc >> 5;
  int row = l - 32;
  float v = (row >= 0 && row < T1N) ? x[((size_t)(b * T1N + row)) * IC + c] : 0.f;
  xt[gid] = (_Float16)v;
}

// ---------------------------------------------------------------------------
// Main fused kernel. One workgroup = one batch b, 16 consecutive t2 positions.
// ---------------------------------------------------------------------------
__global__ __launch_bounds__(256) void cconv_main(const _Float16* __restrict__ xt,
                                                  const float* __restrict__ idxf,
                                                  const float* __restrict__ bias,
                                                  const unsigned* __restrict__ Wh,
                                                  float* __restrict__ out) {
  __shared__ _Float16 Sh[TTILE][64];                                // 2 KB sinc filters
  __shared__ __attribute__((aligned(16))) _Float16 yh[TTILE][1024]; // 32 KB y (one m-half)
  __shared__ __attribute__((aligned(16))) _Float16 xw[4][32][72];   // 18 KB window group
  __shared__ float red[4][256];                                     // 4 KB K-split reduce
  __shared__ int   ipos[TTILE];
  __shared__ float frac[TTILE];

  const int tid  = threadIdx.x;
  const int lane = tid & 31;
  const int wave = tid >> 5;
  const int b    = blockIdx.x >> 5;            // 4 batches * 32 tiles
  const int t0   = (blockIdx.x & 31) * TTILE;

  // Warm L2 with a slice of the packed weight image (global_prefetch_b8).
  __builtin_prefetch(Wh + ((size_t)tid << 8), 0, 0);

  if (tid < TTILE) {
    float v  = idxf[(size_t)b * T2N + t0 + tid];
    float fl = floorf(v);
    ipos[tid] = (int)fl;
    frac[tid] = v - fl;
  }
  __syncthreads();

  // Phase 1: periodic-sinc fractional-delay filter, closed form (Dirichlet):
  //   S[d] = ( sin(31.5*th)/sin(0.5*th) + cos(32*th) ) / 64,  th = pi*(f+d)/32
  {
    int p  = tid >> 4;
    int d0 = (tid & 15) * 4;
    float f = frac[p];
#pragma unroll
    for (int q = 0; q < 4; ++q) {
      float d   = (float)(d0 + q);
      float th  = (f + d) * 0.09817477042f;    // pi/32
      float den = __sinf(0.5f * th);
      float num = __sinf(31.5f * th);
      float ratio = (fabsf(den) > 1e-6f) ? (num / den) : 63.0f;
      float s = (ratio + __cosf(32.0f * th)) * 0.015625f;
      Sh[p][d0 + q] = (_Float16)s;
    }
  }
  __syncthreads();

  v8f acc4 = {0.f, 0.f, 0.f, 0.f, 0.f, 0.f, 0.f, 0.f};  // stage-4 persistent acc
  const int otile = wave & 3;
  const int khalf = wave >> 2;
  const unsigned* yw = (const unsigned*)&yh[0][0];
  const unsigned xwBase = (unsigned)(uintptr_t)&xw[0][0][0];  // LDS byte offset
  const _Float16* xtb = xt + (size_t)b * IC * XTL;

  for (int pass = 0; pass < 2; ++pass) {
    // ---- Stage 3: y[t, m, c], m in [pass*32, pass*32+32), f16 WMMA ---------
    for (int grp = 0; grp < 4; ++grp) {
      // Async-stage 4 position windows (4 x 32ch x 72 halves = 1152 x 16B)
      // through the gfx1250 async global->LDS path (tracked by ASYNCcnt).
      for (int q = 0; q < 5; ++q) {
        int g = q * 256 + tid;
        if (g < 1152) {
          int pl = g / 288;                    // 288 = 32 ch * 9 granules
          int rc = g - pl * 288;
          int c  = rc / 9;
          int gr = rc - c * 9;
          int it = ipos[grp * 4 + pl];
          int sa = it & ~7;                    // 16B-aligned window start
          const _Float16* gp = xtb + (size_t)c * XTL + sa + gr * 8;
          unsigned lds = xwBase + (unsigned)(((pl * 32 + c) * 72 + gr * 8) * 2);
          unsigned long long ga = (unsigned long long)(uintptr_t)gp;
          asm volatile("global_load_async_to_lds_b128 %0, %1, off"
                       :: "v"(lds), "v"(ga) : "memory");
        }
      }
      asm volatile("s_wait_asynccnt 0" ::: "memory");
      __syncthreads();

      // 16 tiles (4 pos x 2 mt x 2 ct) over 8 waves -> 2 tiles each
      for (int ii = 0; ii < 2; ++ii) {
        int combo = wave * 2 + ii;
        int pl = combo >> 2;
        int mt = (combo >> 1) & 1;
        int ct = combo & 1;
        int p  = grp * 4 + pl;
        int m0 = pass * 32 + mt * 16;
        int c0 = ct * 16;
        int base = ipos[p] & 7;                // misalignment within staged row
        v8f acc = {0.f, 0.f, 0.f, 0.f, 0.f, 0.f, 0.f, 0.f};
        for (int kc = 0; kc < 2; ++kc) {
          v16h Af, Bf;
          int Mrow = m0 + (lane & 15);
          int cidx = c0 + (lane & 15);
#pragma unroll
          for (int e = 0; e < 16; ++e) {
            int j = e >> 1, po = e & 1;
            // A (16x32 f16 layout): S_t[(k - M) mod 64], circulant matrix
            int Ka  = ((lane >> 4) << 4) + (j << 1) + po;
            int kgA = kc * 32 + Ka;
            Af[e] = Sh[p][(kgA - Mrow) & 63];
            // B (32x16 f16 layout): window from LDS, no bounds check / no cvt
            int Kb  = ((lane >> 4) << 4) + (j << 1) + po;
            Bf[e] = xw[pl][cidx][base + kc * 32 + Kb];
          }
          acc = __builtin_amdgcn_wmma_f32_16x16x32_f16(false, Af, false, Bf,
                                                       (short)0, acc, false, false);
        }
        int cc = c0 + (lane & 15);
#pragma unroll
        for (int v = 0; v < 8; ++v) {
          int mloc = mt * 16 + v + ((lane >> 4) << 3);   // C/D layout row
          yh[p][mloc * 32 + cc] = (_Float16)acc[v];
        }
      }
      __syncthreads();   // xw reused next group; yh complete after last group
    }

    // ---- Stage 4: out(16x64) += Y(16x1024_pass) * W^T, f16 WMMA -----------
    for (int kc4 = khalf * 16; kc4 < khalf * 16 + 16; ++kc4) {
      int kcg = pass * 32 + kc4;
      v16h Af, Bf;
      int trow = lane & 15;
#pragma unroll
      for (int j = 0; j < 8; ++j) {
        int Kb = ((lane >> 4) << 4) + (j << 1);
        HU cv;
        cv.u = yw[(trow * 1024 + kc4 * 32 + Kb) >> 1];   // ds_load_b32 (vectorized)
        Af[2 * j]     = cv.h[0];
        Af[2 * j + 1] = cv.h[1];
      }
      const uint4* wp4 =
          (const uint4*)(Wh + (((size_t)(otile * 64 + kcg) * 32 + lane) << 3));
      uint4 lo = wp4[0], hi = wp4[1];                    // 2x global_load_b128
      {
        HU cv;
        cv.u = lo.x; Bf[0]  = cv.h[0]; Bf[1]  = cv.h[1];
        cv.u = lo.y; Bf[2]  = cv.h[0]; Bf[3]  = cv.h[1];
        cv.u = lo.z; Bf[4]  = cv.h[0]; Bf[5]  = cv.h[1];
        cv.u = lo.w; Bf[6]  = cv.h[0]; Bf[7]  = cv.h[1];
        cv.u = hi.x; Bf[8]  = cv.h[0]; Bf[9]  = cv.h[1];
        cv.u = hi.y; Bf[10] = cv.h[0]; Bf[11] = cv.h[1];
        cv.u = hi.z; Bf[12] = cv.h[0]; Bf[13] = cv.h[1];
        cv.u = hi.w; Bf[14] = cv.h[0]; Bf[15] = cv.h[1];
      }
      acc4 = __builtin_amdgcn_wmma_f32_16x16x32_f16(false, Af, false, Bf,
                                                    (short)0, acc4, false, false);
    }
    __syncthreads();   // yh is overwritten next pass
  }

  // ---- Cross-wave K reduction + bias + store --------------------------------
  if (khalf == 1) {
#pragma unroll
    for (int v = 0; v < 8; ++v) red[otile][v * 32 + lane] = acc4[v];
  }
  __syncthreads();
  if (khalf == 0) {
    int o = otile * 16 + (lane & 15);
    float bo = bias[o];
#pragma unroll
    for (int v = 0; v < 8; ++v) {
      int t = v + ((lane >> 4) << 3);
      out[((size_t)(b * T2N + t0 + t)) * OC + o] =
          acc4[v] + red[otile][v * 32 + lane] + bo;
    }
  }
}

// ---------------------------------------------------------------------------
extern "C" void kernel_launch(void* const* d_in, const int* in_sizes, int n_in,
                              void* d_out, int out_size, void* d_ws, size_t ws_size,
                              hipStream_t stream) {
  const float* x    = (const float*)d_in[0];   // (4, 4096, 32)
  const float* idx  = (const float*)d_in[1];   // (4, 512)
  const float* kern = (const float*)d_in[2];   // (64, 32, 64)
  const float* bias = (const float*)d_in[3];   // (64,)
  float* out = (float*)d_out;                  // (4, 512, 64)

  unsigned*  Wh = (unsigned*)d_ws;                         // 256 KB packed f16 W
  _Float16*  xt = (_Float16*)((char*)d_ws + 65536 * 4);    // ~1.02 MB padded f16 x

  cconv_prep_w<<<256, 256, 0, stream>>>(kern, Wh);
  cconv_prep_x<<<(BN * IC * XTL) / 256, 256, 0, stream>>>(x, xt);
  cconv_main<<<BN * (T2N / TTILE), 256, 0, stream>>>(xt, idx, bias, Wh, out);
}
